// DynamicPostionalBias_23819888623837
// MI455X (gfx1250) — compile-verified, matrix operand order checked by reading, not run on hardware
//
#include <hip/hip_runtime.h>
#include <stdint.h>

// ---- problem constants (from reference) ----
#define B_  2
#define H_  16
#define S_  2048
#define D_  64
#define J_  4097          // 2*2048+1 relative slots

// ---- workspace layout (bytes), all 256B aligned ----
#define QHI_OFF  (0u)
#define QLO_OFF  (8388608u)     // B*H*S*D * 2B = 8 MB
#define EHI_OFF  (16777216u)
#define ELO_OFF  (25167872u)    // EHI + H*J*D*2B (8,390,656 -> padded)

typedef __attribute__((ext_vector_type(16))) __bf16 v16bf;
typedef __attribute__((ext_vector_type(8)))  float  v8f;

union V16 { v16bf v; uint4 u[2]; };
union BF4 { __bf16 b[4]; uint2 u; };

// ------------------------------------------------------------------
// Kernel 1: split Q (f32) into bf16 hi + bf16 lo residual, elementwise.
// ------------------------------------------------------------------
__global__ __launch_bounds__(256) void q_split_kernel(
    const float4* __restrict__ q, __bf16* __restrict__ qhi, __bf16* __restrict__ qlo)
{
    const int idx = blockIdx.x * 256 + threadIdx.x;   // covers (B*H*S*D)/4
    float4 x = q[idx];
    BF4 hi, lo;
    float xs[4] = {x.x, x.y, x.z, x.w};
    #pragma unroll
    for (int i = 0; i < 4; ++i) {
        __bf16 h = (__bf16)xs[i];
        hi.b[i] = h;
        lo.b[i] = (__bf16)(xs[i] - (float)h);
    }
    *(uint2*)(qhi + 4 * (size_t)idx) = hi.u;
    *(uint2*)(qlo + 4 * (size_t)idx) = lo.u;
}

// ------------------------------------------------------------------
// Kernel 2: transpose E [H][D][J] -> [H][J][D] and split into bf16 hi/lo.
// ------------------------------------------------------------------
__global__ __launch_bounds__(256) void e_split_t_kernel(
    const float* __restrict__ e, __bf16* __restrict__ ehi, __bf16* __restrict__ elo)
{
    const int col = blockIdx.x * 256 + threadIdx.x;   // h*J + j
    if (col >= H_ * J_) return;
    const int h = col / J_;
    const int j = col - h * J_;
    const float* src = e + (size_t)h * D_ * J_ + j;
    __bf16* dh = ehi + (size_t)col * D_;
    __bf16* dl = elo + (size_t)col * D_;
    #pragma unroll 8
    for (int d = 0; d < D_; ++d) {
        float x = src[(size_t)d * J_];
        __bf16 hi = (__bf16)x;
        dh[d] = hi;
        dl[d] = (__bf16)(x - (float)hi);
    }
}

// ------------------------------------------------------------------
// Shared tile body: 8x global_load_b128 + 6x v_wmma_f32_16x16x32_bf16
// (bf16x3: q*e ~= qh*eh + qh*el + ql*eh, f32 accumulate).
// pHi/pLo already include the per-lane column (+Mn*D) and +hK offsets.
// ------------------------------------------------------------------
__device__ __forceinline__ v8f compute_tile(
    const __bf16* __restrict__ pHi, const __bf16* __restrict__ pLo,
    const V16 (&ahi)[2], const V16 (&alo)[2])
{
    V16 bhi[2], blo[2];
    #pragma unroll
    for (int kh = 0; kh < 2; ++kh) {
        bhi[kh].u[0] = *(const uint4*)(pHi + 32 * kh);
        bhi[kh].u[1] = *(const uint4*)(pHi + 32 * kh + 16);
        blo[kh].u[0] = *(const uint4*)(pLo + 32 * kh);
        blo[kh].u[1] = *(const uint4*)(pLo + 32 * kh + 16);
    }
    v8f c = {};
    #pragma unroll
    for (int kh = 0; kh < 2; ++kh) {
        c = __builtin_amdgcn_wmma_f32_16x16x32_bf16(false, ahi[kh].v, false, bhi[kh].v, (short)0, c, false, false);
        c = __builtin_amdgcn_wmma_f32_16x16x32_bf16(false, ahi[kh].v, false, blo[kh].v, (short)0, c, false, false);
        c = __builtin_amdgcn_wmma_f32_16x16x32_bf16(false, alo[kh].v, false, bhi[kh].v, (short)0, c, false, false);
    }
    return c;
}

// ------------------------------------------------------------------
// Main kernel: banded GEMM with diagonal scatter out[n,h,i,k], k=j+i-2048.
// k = 16t + M + N - 16, so only tiles t=0 and t=128 need bounds guards.
// Interior tiles (t=1..127) run in a branch-free loop with constant
// pointer increments; wave 7 (15 interior tiles) also does both edges.
// ------------------------------------------------------------------
__global__ __launch_bounds__(256) void rel_bias_wmma_kernel(
    const __bf16* __restrict__ qhi, const __bf16* __restrict__ qlo,
    const __bf16* __restrict__ ehi, const __bf16* __restrict__ elo,
    float* __restrict__ out)
{
    const int i0    = blockIdx.x << 4;       // row tile base (0..2032)
    const int h     = blockIdx.y;
    const int n     = blockIdx.z;
    const int wave  = threadIdx.x >> 5;
    const int lane  = threadIdx.x & 31;
    const int Mn    = lane & 15;             // A: row M within tile; B: col N
    const int hK    = (lane >> 4) << 3;      // 0 or 8 (K sub-block per lane half)
    const int Mbase = (lane >> 4) << 3;      // C/D: M = v + 8*(lane>=16)

    // ---- A operands (Q tile), loaded once, reused for all tiles ----
    const size_t qrow = (((size_t)n * H_ + h) * S_ + (i0 + Mn)) * D_;
    V16 ahi[2], alo[2];
    #pragma unroll
    for (int kh = 0; kh < 2; ++kh) {
        size_t b0 = qrow + 32 * kh + hK;
        ahi[kh].u[0] = *(const uint4*)(qhi + b0);
        ahi[kh].u[1] = *(const uint4*)(qhi + b0 + 16);
        alo[kh].u[0] = *(const uint4*)(qlo + b0);
        alo[kh].u[1] = *(const uint4*)(qlo + b0 + 16);
    }

    const int    jstart = 2032 - i0;                       // band start (tile aligned)
    const size_t outNH  = ((size_t)n * H_ + h) * S_ * S_;
    // base of the v=0 store for tile t: outNH + (i0+Mbase)*S + 16*(t-1) + Mbase + Mn
    float* const outRow = out + outNH + (size_t)(i0 + Mbase) * S_ + (Mbase + Mn);

    // ---- interior tiles: t = wave+1, wave+9, ... <= 127 (branch-free) ----
    {
        const int t0 = wave + 1;
        const size_t ecol0 = ((size_t)h * J_ + (jstart + (t0 << 4) + Mn)) * D_ + hK;
        const __bf16* pHi = ehi + ecol0;
        const __bf16* pLo = elo + ecol0;
        float*        p   = outRow + (size_t)(t0 - 1) * 16;
        for (int t = t0; t <= 127; t += 8) {
            v8f c = compute_tile(pHi, pLo, ahi, alo);
            #pragma unroll
            for (int v = 0; v < 8; ++v) {
                p[(size_t)v * (S_ + 1)] = c[v];   // imm offset 8196*v
            }
            pHi += 128 * D_;                      // +8 tiles = 128 columns
            pLo += 128 * D_;
            p   += 128;                           // +8 tiles * 16 in k
        }
    }

    // ---- edge tiles t=0 and t=128 (guarded), done by wave 7 ----
    if (wave == 7) {
        #pragma unroll
        for (int tt = 0; tt < 2; ++tt) {
            const int t     = tt * 128;           // 0 or 128
            const int jbase = jstart + (t << 4);
            const size_t ecol = ((size_t)h * J_ + (jbase + Mn)) * D_ + hK;
            v8f c = compute_tile(ehi + ecol, elo + ecol, ahi, alo);
            #pragma unroll
            for (int v = 0; v < 8; ++v) {
                const int i = i0 + Mbase + v;
                const int k = jbase + Mn + i - 2048;
                if ((unsigned)k < (unsigned)S_) {
                    out[outNH + (size_t)i * S_ + k] = c[v];
                }
            }
        }
    }
}

// ------------------------------------------------------------------
extern "C" void kernel_launch(void* const* d_in, const int* in_sizes, int n_in,
                              void* d_out, int out_size, void* d_ws, size_t ws_size,
                              hipStream_t stream)
{
    const float* q = (const float*)d_in[0];   // [B,H,S,D] f32
    const float* e = (const float*)d_in[1];   // [H,D,J]   f32
    // d_in[2] = sequence_length (== S_), compile-time constant here.
    float* out = (float*)d_out;               // [B,H,S,S] f32

    char* ws = (char*)d_ws;
    __bf16* qhi = (__bf16*)(ws + QHI_OFF);
    __bf16* qlo = (__bf16*)(ws + QLO_OFF);
    __bf16* ehi = (__bf16*)(ws + EHI_OFF);
    __bf16* elo = (__bf16*)(ws + ELO_OFF);

    // 1) split Q: B*H*S*D = 4,194,304 elems -> /4 per thread -> 4096 blocks
    q_split_kernel<<<4096, 256, 0, stream>>>((const float4*)q, qhi, qlo);

    // 2) transpose + split E: H*J = 65,552 columns
    e_split_t_kernel<<<(H_ * J_ + 255) / 256, 256, 0, stream>>>(e, ehi, elo);

    // 3) banded WMMA GEMM + diagonal scatter
    dim3 grid(S_ / 16, H_, B_);
    rel_bias_wmma_kernel<<<grid, 256, 0, stream>>>(qhi, qlo, ehi, elo, out);
}
